// LumenLinear_40072044872037
// MI455X (gfx1250) — 8-bit matrix operand order repaired and checked by reading, compiled, not run on hardware
//
#include <hip/hip_runtime.h>
#include <hip/hip_bf16.h>

typedef __attribute__((ext_vector_type(16))) int   v16i;
typedef __attribute__((ext_vector_type(4)))  int   v4i;
typedef __attribute__((ext_vector_type(8)))  float v8f;

#define FP8_MAX 448.0f
#define QEPS 1e-12f

#if defined(__gfx1250__) && __has_builtin(__builtin_amdgcn_global_load_async_to_lds_b128)
#define HAVE_ASYNC_LDS 1
typedef __attribute__((address_space(1))) v4i g_v4i;
typedef __attribute__((address_space(3))) v4i l_v4i;
#else
#define HAVE_ASYNC_LDS 0
#endif

#if HAVE_ASYNC_LDS
#if __has_builtin(__builtin_amdgcn_s_wait_asynccnt)
#define WAIT_ASYNC(n) __builtin_amdgcn_s_wait_asynccnt(n)
#else
#define WAIT_ASYNC(n) asm volatile("s_wait_asynccnt " #n ::: "memory")
#endif
#else
#define WAIT_ASYNC(n) do {} while (0)
#endif

__device__ __forceinline__ unsigned int f32_to_e4m3(float f) {
    unsigned int sign = (__float_as_uint(f) >> 24) & 0x80u;
    float a = fminf(fabsf(f), FP8_MAX);
    unsigned int r;
    if (a < 0.015625f) {
        r = (unsigned int)__float2int_rn(a * 512.0f);
    } else {
        unsigned int x = __float_as_uint(a);
        x += 0x7FFFFu + ((x >> 20) & 1u);
        unsigned int e8 = ((x >> 23) & 0xFFu) - 120u;
        r = (e8 << 3) | ((x >> 20) & 7u);
    }
    return r | sign;
}

__global__ __launch_bounds__(256) void qx_kernel(
    const float* __restrict__ x, unsigned char* __restrict__ qx,
    float* __restrict__ xs, int M, int K, int KB) {
    const int lane = threadIdx.x & 31;
    const int wid  = (blockIdx.x * blockDim.x + threadIdx.x) >> 5;
    const int row  = wid / KB;
    const int kb   = wid - row * KB;

    const float4 v = *(const float4*)(x + (size_t)row * K + (kb << 7) + lane * 4);
    float amax = fmaxf(fmaxf(fabsf(v.x), fabsf(v.y)), fmaxf(fabsf(v.z), fabsf(v.w)));
    #pragma unroll
    for (int off = 16; off > 0; off >>= 1)
        amax = fmaxf(amax, __shfl_xor(amax, off, 32));

    const float scale = fmaxf(amax, QEPS) * (1.0f / FP8_MAX);
    const float inv   = 1.0f / scale;
    const unsigned int b = f32_to_e4m3(v.x * inv)
                         | (f32_to_e4m3(v.y * inv) << 8)
                         | (f32_to_e4m3(v.z * inv) << 16)
                         | (f32_to_e4m3(v.w * inv) << 24);
    *(unsigned int*)(qx + (size_t)row * K + (kb << 7) + lane * 4) = b;
    if (lane == 0) xs[(size_t)kb * M + row] = scale;
}

__global__ __launch_bounds__(256) void qw_kernel(
    const float* __restrict__ w, unsigned char* __restrict__ qw,
    float* __restrict__ wsc, int K, int KB) {
    __shared__ float red[256];
    const int nb = blockIdx.y;
    const int kb = blockIdx.x;
    const int t  = threadIdx.x;
    const int row  = (nb << 7) + (t >> 1);
    const int col0 = (kb << 7) + ((t & 1) << 6);

    const float4* p = (const float4*)(w + (size_t)row * K + col0);
    float4 v[16];
    float amax = 0.0f;
    #pragma unroll
    for (int i = 0; i < 16; ++i) {
        v[i] = p[i];
        amax = fmaxf(amax, fmaxf(fmaxf(fabsf(v[i].x), fabsf(v[i].y)),
                                 fmaxf(fabsf(v[i].z), fabsf(v[i].w))));
    }
    red[t] = amax;
    __syncthreads();
    #pragma unroll
    for (int s = 128; s > 0; s >>= 1) {
        if (t < s) red[t] = fmaxf(red[t], red[t + s]);
        __syncthreads();
    }
    const float scale = fmaxf(red[0], QEPS) * (1.0f / FP8_MAX);
    if (t == 0) wsc[(size_t)nb * KB + kb] = scale;
    const float inv = 1.0f / scale;

    unsigned int* q = (unsigned int*)(qw + (size_t)row * K + col0);
    #pragma unroll
    for (int i = 0; i < 16; ++i) {
        q[i] = f32_to_e4m3(v[i].x * inv)
             | (f32_to_e4m3(v[i].y * inv) << 8)
             | (f32_to_e4m3(v[i].z * inv) << 16)
             | (f32_to_e4m3(v[i].w * inv) << 24);
    }
}

#define LDS_PITCH 144

__global__ __launch_bounds__(256) void gemm_fp8_kernel(
    const unsigned char* __restrict__ qx, const unsigned char* __restrict__ qw,
    const float* __restrict__ xs, const float* __restrict__ wsc,
    const float* __restrict__ bias, float* __restrict__ out,
    int M, int N, int K, int KB) {
    __shared__ unsigned char bsm[2][128 * LDS_PITCH];

    const int tid  = threadIdx.x;
    const int lane = tid & 31;
    const int wave = tid >> 5;
    const int bn   = blockIdx.x;
    const int bm   = blockIdx.y;
    const int m0   = (bm << 7) + (wave << 4);
    const int n0   = bn << 7;

    const int half = (lane >> 4) & 1;
    const int l15  = lane & 15;

    const int srow  = tid >> 1;
    const int sboff = (tid & 1) << 6;
    const unsigned char* gstage = qw + (size_t)(n0 + srow) * K + sboff;
    unsigned char* lstage0 = &bsm[0][srow * LDS_PITCH + sboff];
    unsigned char* lstage1 = &bsm[1][srow * LDS_PITCH + sboff];

    const unsigned char* ap_base = qx + (size_t)(m0 + l15) * K + (half ? 16 : 0);
    const float* xs_base = xs + m0 + (half << 3);
    const float* ws_base = wsc + (size_t)bn * KB;

    auto stage = [&](unsigned char* l, int kb) {
        const unsigned char* g = gstage + (kb << 7);
        #pragma unroll
        for (int i = 0; i < 4; ++i) {
#if HAVE_ASYNC_LDS
            __builtin_amdgcn_global_load_async_to_lds_b128(
                (g_v4i*)(g + 16 * i), (l_v4i*)(l + 16 * i), 0, 0);
#else
            *(int4*)(l + 16 * i) = *(const int4*)(g + 16 * i);
#endif
        }
    };

    v8f C[8];
    #pragma unroll
    for (int t = 0; t < 8; ++t)
        #pragma unroll
        for (int i = 0; i < 8; ++i) C[t][i] = 0.0f;

    stage(lstage0, 0);
    if (KB > 1) stage(lstage1, 1);

    for (int kb = 0; kb < KB; ++kb) {
        const int cur = kb & 1;
        const int k0  = kb << 7;

        WAIT_ASYNC(4);
        __syncthreads();

        v16i A;
        const unsigned char* ap = ap_base + k0;
        #pragma unroll
        for (int i = 0; i < 4; ++i) {
            const int4 d = *(const int4*)(ap + 32 * i);
            A[4 * i]     = d.x;
            A[4 * i + 1] = d.y;
            A[4 * i + 2] = d.z;
            A[4 * i + 3] = d.w;
        }

        const float4 sa0 = *(const float4*)(xs_base + (size_t)kb * M);
        const float4 sa1 = *(const float4*)(xs_base + (size_t)kb * M + 4);
        float sa[8] = {sa0.x, sa0.y, sa0.z, sa0.w, sa1.x, sa1.y, sa1.z, sa1.w};
        const float sw = ws_base[kb];

        const unsigned char* bl = &bsm[cur][half << 4];
        #pragma unroll
        for (int t = 0; t < 8; ++t) {
            v16i B;
            const unsigned char* bp = bl + (t * 16 + l15) * LDS_PITCH;
            #pragma unroll
            for (int g = 0; g < 4; ++g) {
                const int4 d = *(const int4*)(bp + 32 * g);
                B[4 * g]     = d.x;
                B[4 * g + 1] = d.y;
                B[4 * g + 2] = d.z;
                B[4 * g + 3] = d.w;
            }
            v8f z = {0.f, 0.f, 0.f, 0.f, 0.f, 0.f, 0.f, 0.f};
            v8f d8 = __builtin_amdgcn_wmma_f32_16x16x128_fp8_fp8(
                A, B, (short)0, z, false, false);
            #pragma unroll
            for (int i = 0; i < 8; ++i)
                C[t][i] += (sa[i] * sw) * d8[i];
        }

        __syncthreads();
        if (kb + 2 < KB) stage(cur ? lstage1 : lstage0, kb + 2);
    }

    #pragma unroll
    for (int t = 0; t < 8; ++t) {
        const int n  = n0 + t * 16 + l15;
        const float bz = bias[n];
        #pragma unroll
        for (int i = 0; i < 8; ++i) {
            const int m = m0 + (half << 3) + i;
            out[(size_t)m * N + n] = C[t][i] + bz;
        }
    }
}

extern "C" void kernel_launch(void* const* d_in, const int* in_sizes, int n_in,
                              void* d_out, int out_size, void* d_ws, size_t ws_size,
                              hipStream_t stream) {
    const float* x    = (const float*)d_in[0];
    const float* w    = (const float*)d_in[1];
    const float* bias = (const float*)d_in[2];

    const int N = in_sizes[2];
    const int K = (int)((long long)in_sizes[1] / N);
    const int M = (int)((long long)in_sizes[0] / K);
    const int KB = K >> 7;
    const int NB = N >> 7;
    const int MB = M >> 7;

    unsigned char* qx = (unsigned char*)d_ws;
    unsigned char* qw = qx + (size_t)M * K;
    float* xs  = (float*)(qw + (size_t)N * K);
    float* wsc = xs + (size_t)M * KB;

    float* out = (float*)d_out;

    {
        const int waves  = M * KB;
        const int blocks = waves / 8;
        qx_kernel<<<blocks, 256, 0, stream>>>(x, qx, xs, M, K, KB);
    }
    {
        dim3 grid(KB, NB);
        qw_kernel<<<grid, 256, 0, stream>>>(w, qw, wsc, K, KB);
    }
    {
        dim3 grid(NB, MB);
        gemm_fp8_kernel<<<grid, 256, 0, stream>>>(qx, qw, xs, wsc, bias, out,
                                                  M, N, K, KB);
    }
}
